// CausalSelfAttention_76295799046279
// MI455X (gfx1250) — matrix operand order repaired and checked by reading, not compiled, not run on hardware
//
#include <hip/hip_runtime.h>
#include <hip/hip_bf16.h>

// ---------------- CDNA5 WMMA types/helpers ----------------
typedef __attribute__((ext_vector_type(16))) __bf16          v16bf;
typedef __attribute__((ext_vector_type(8)))  float           v8f;
typedef __attribute__((ext_vector_type(8)))  unsigned short  us8;
typedef __attribute__((ext_vector_type(4)))  unsigned int    u32x4;
typedef __attribute__((ext_vector_type(8)))  int             i32x8;
typedef __attribute__((ext_vector_type(4)))  int             i32x4;

union Frag32B { us8 h[2]; v16bf v; };

static __device__ __forceinline__ v16bf make_frag(const unsigned short* p0,
                                                  const unsigned short* p1) {
  Frag32B f;
  f.h[0] = *(const us8*)p0;
  f.h[1] = *(const us8*)p1;
  return f.v;
}

static __device__ __forceinline__ v8f wmma_bf16(v16bf a, v16bf b, v8f c) {
  return __builtin_amdgcn_wmma_f32_16x16x32_bf16(
      false, a, false, b, (short)0, c, false, false);
}

static __device__ __forceinline__ unsigned short f2bf(float x) {
  unsigned u = __float_as_uint(x);
  unsigned r = (u + 0x7FFFu + ((u >> 16) & 1u)) >> 16;  // RNE
  return (unsigned short)r;
}

// ---- gfx1250 async memory->LDS (ASYNCcnt path), used in attention ----
static __device__ __forceinline__ void async_g2l_b128(unsigned short* lds,
                                                      const unsigned short* g) {
  unsigned l = (unsigned)(unsigned long long)(void*)lds;  // low 32 bits = LDS offset
  unsigned long long ga = (unsigned long long)(const void*)g;
  asm volatile("global_load_async_to_lds_b128 %0, %1, off"
               :: "v"(l), "v"(ga) : "memory");
}
static __device__ __forceinline__ void wait_async0() {
  asm volatile("s_wait_asynccnt 0" ::: "memory");
}

// ---- gfx1250 Tensor Data Mover: 2-D tile global -> LDS (TENSORcnt path) ----
// Descriptor per CDNA5 ISA ch.8: group0 = {count/lds_addr/global_addr/type},
// group1 = {data_size, tensor dims, tile dims, dim0 stride}. 2-D: higher groups zero.
// This toolchain's builtin takes 6 args (g0, g1, g2, g3, g4, cpol).
static __device__ __forceinline__ void tdm_load_2d(unsigned lds_byte_off,
                                                   const void* gaddr,
                                                   int tile_rows, int tile_cols_elems,
                                                   int row_stride_elems,
                                                   int tensor_rows) {
  unsigned long long ga = (unsigned long long)gaddr;
  u32x4 g0;
  g0[0] = 1u;                                            // count=1, no gather
  g0[1] = lds_byte_off;                                  // lds_addr
  g0[2] = (unsigned)(ga & 0xFFFFFFFFu);                  // global_addr[31:0]
  g0[3] = (unsigned)((ga >> 32) & 0x1FFFFFFu) | (2u << 30);  // addr[56:32] | type=2
  i32x8 g1;
  g1[0] = (1 << 16);                                     // data_size=2B; mask/flags=0
  g1[1] = (row_stride_elems & 0xFFFF) << 16;             // tensor_dim0[15:0] @63:48
  g1[2] = ((row_stride_elems >> 16) & 0xFFFF)            // tensor_dim0[31:16]
        | ((tensor_rows & 0xFFFF) << 16);                // tensor_dim1[15:0]
  g1[3] = ((tensor_rows >> 16) & 0xFFFF)                 // tensor_dim1[31:16]
        | (tile_cols_elems << 16);                       // tile_dim0
  g1[4] = tile_rows & 0xFFFF;                            // tile_dim1 (tile_dim2=0)
  g1[5] = row_stride_elems;                              // tensor_dim0_stride[31:0]
  g1[6] = 0;
  g1[7] = 0;
  i32x4 z4 = {0, 0, 0, 0};
  i32x8 z8 = {0, 0, 0, 0, 0, 0, 0, 0};
  __builtin_amdgcn_tensor_load_to_lds(g0, g1, z4, z4, z8, 0);
}
static __device__ __forceinline__ void wait_tensor0() {
  __builtin_amdgcn_s_wait_tensorcnt(0);
}

// ---------------- problem constants ----------------
#define BB  2
#define TT  2048
#define CC  1024
#define HH  16
#define DD  64
#define MM  (BB * TT)          // 4096 tokens

// ---------------- fp32 -> bf16 cast ----------------
__global__ __launch_bounds__(256) void cast_f32_bf16(const float* __restrict__ in,
                                                     unsigned short* __restrict__ out,
                                                     int n) {
  int i = blockIdx.x * 256 + threadIdx.x;
  if (i < n) out[i] = f2bf(in[i]);
}

// ---------------- GEMM: Y[M,N] = A[M,K](bf16) @ W[N,K]^T(bf16) + bias ----------------
// Block tile 128x128, 8 waves (4Mx2N), wave tile 32x64. K-step 64 -> 16 WMMA per
// barrier pair. Double-buffered LDS staged by the Tensor Data Mover (wave 0 issues
// one tensor_load_to_lds per operand tile; TENSORcnt synchronized).
// MODE 0: bf16 [B,H,T,D] (Q,K)   MODE 1: bf16 [B,H,D,T] (V^T)   MODE 2: fp32 [M,N]
template <int MODE>
__global__ __launch_bounds__(256) void gemm_bf16(const unsigned short* __restrict__ A,
                                                 const unsigned short* __restrict__ W,
                                                 const float* __restrict__ bias,
                                                 void* __restrict__ out,
                                                 int K, int Ntot, int Mtot) {
  __shared__ __align__(16) unsigned short lsA[2 * 128 * 64];   // 32 KB
  __shared__ __align__(16) unsigned short lsW[2 * 128 * 64];   // 32 KB

  const int tid    = threadIdx.x;
  const int lane   = tid & 31;
  const int w      = tid >> 5;
  const int lane15 = lane & 15;
  const int laneHi = lane >> 4;
  const int mBase  = blockIdx.y * 128;
  const int nBase  = blockIdx.x * 128;
  const int waveM  = (w & 3) * 32;
  const int waveN  = (w >> 2) * 64;

  const unsigned ldsA0 = (unsigned)(unsigned long long)(void*)lsA;
  const unsigned ldsW0 = (unsigned)(unsigned long long)(void*)lsW;

  v8f acc[2][4] = {};

  auto stage = [&](int bi, int k0) {   // wave-uniform; called by wave 0 only
    tdm_load_2d(ldsA0 + bi * 16384, A + (size_t)mBase * K + k0,
                128, 64, K, Mtot);
    tdm_load_2d(ldsW0 + bi * 16384, W + (size_t)nBase * K + k0,
                128, 64, K, Ntot);
  };

  if (w == 0) { stage(0, 0); wait_tensor0(); }
  __syncthreads();

  const int nk = K / 64;
  for (int kb = 0; kb < nk; ++kb) {
    const int cur = kb & 1;
    if (w == 0 && kb + 1 < nk) stage((kb + 1) & 1, (kb + 1) * 64);

    const unsigned short* pa = lsA + cur * 8192;
    const unsigned short* pwt = lsW + cur * 8192;
#pragma unroll
    for (int j = 0; j < 2; ++j) {      // two K=32 chunks
      v16bf afr[2], bfr[4];
#pragma unroll
      for (int mi = 0; mi < 2; ++mi) {
        const unsigned short* p = pa + (waveM + mi * 16 + lane15) * 64 + j * 32 + laneHi * 8;
        afr[mi] = make_frag(p, p + 16);
      }
#pragma unroll
      for (int ni = 0; ni < 4; ++ni) {
        const unsigned short* p = pwt + (waveN + ni * 16 + lane15) * 64 + j * 32 + laneHi * 8;
        bfr[ni] = make_frag(p, p + 16);
      }
#pragma unroll
      for (int mi = 0; mi < 2; ++mi)
#pragma unroll
        for (int ni = 0; ni < 4; ++ni)
          acc[mi][ni] = wmma_bf16(afr[mi], bfr[ni], acc[mi][ni]);
    }

    if (w == 0) wait_tensor0();
    __syncthreads();
  }

  // epilogue: D-layout => vgpr r: lanes0-15 row r, lanes16-31 row r+8; col = lane&15
#pragma unroll
  for (int mi = 0; mi < 2; ++mi) {
#pragma unroll
    for (int ni = 0; ni < 4; ++ni) {
      int coln = nBase + waveN + ni * 16 + lane15;
      float bn = bias[coln];
      int rowb = mBase + waveM + mi * 16 + 8 * laneHi;
#pragma unroll
      for (int r = 0; r < 8; ++r) {
        int gm = rowb + r;
        float val = acc[mi][ni][r] + bn;
        if constexpr (MODE == 2) {
          ((float*)out)[(size_t)gm * Ntot + coln] = val;
        } else {
          int b = gm >> 11, t = gm & (TT - 1);
          int h = coln >> 6, d = coln & (DD - 1);
          size_t idx;
          if constexpr (MODE == 0)
            idx = (((size_t)(b * HH + h) * TT + t) * DD) + d;   // [B,H,T,D]
          else
            idx = (((size_t)(b * HH + h) * DD + d) * TT) + t;   // [B,H,D,T]
          ((unsigned short*)out)[idx] = f2bf(val);
        }
      }
    }
  }
}

// ---------------- flash attention ----------------
// q,k: [B,H,T,D] bf16; vT: [B,H,D,T] bf16 -> yatt: [B,T,C] bf16
// k-tile 64, double-buffered async-to-LDS staging, causal tile skipping.
__global__ __launch_bounds__(256) void attn_kernel(const unsigned short* __restrict__ q,
                                                   const unsigned short* __restrict__ kk,
                                                   const unsigned short* __restrict__ vT,
                                                   unsigned short* __restrict__ yatt) {
  __shared__ __align__(16) unsigned short lsK[2 * 64 * 64];
  __shared__ __align__(16) unsigned short lsV[2 * 64 * 64];
  __shared__ __align__(16) unsigned short lsP[8 * 16 * 64];

  const int tid    = threadIdx.x;
  const int lane   = tid & 31;
  const int w      = tid >> 5;
  const int lane15 = lane & 15;
  const int laneHi = lane >> 4;
  const int bh     = blockIdx.y;
  const int b      = bh >> 4;
  const int h      = bh & 15;
  const int qBase  = blockIdx.x * 128;
  const int qRow0  = qBase + w * 16;

  const float NEG_INF = -__builtin_inff();
  const float cs = 0.18033688011112042f;   // (1/sqrt(64)) * log2(e)

  v16bf qf[2];
  {
    size_t base = ((size_t)bh * TT + qRow0 + lane15) * DD;
#pragma unroll
    for (int j = 0; j < 2; ++j) {
      const unsigned short* p = q + base + j * 32 + laneHi * 8;
      qf[j] = make_frag(p, p + 16);
    }
  }

  v8f o[4] = {};
  float mst[8], lst[8];
#pragma unroll
  for (int r = 0; r < 8; ++r) { mst[r] = NEG_INF; lst[r] = 0.f; }

  unsigned short* pw = lsP + w * (16 * 64);

  const int srow = tid >> 2;               // 0..63
  const int scol = (tid & 3) * 16;

  auto stage = [&](int bi, int tk) {
    const unsigned short* gk = kk + ((size_t)bh * TT + tk + srow) * DD + scol;
    unsigned short*       dk = lsK + bi * 4096 + srow * 64 + scol;
    async_g2l_b128(dk,     gk);
    async_g2l_b128(dk + 8, gk + 8);
    const unsigned short* gv = vT + ((size_t)bh * DD + srow) * TT + tk + scol;
    unsigned short*       dv = lsV + bi * 4096 + srow * 64 + scol;
    async_g2l_b128(dv,     gv);
    async_g2l_b128(dv + 8, gv + 8);
  };

  const int kEnd   = qBase + 128;
  const int ntiles = kEnd / 64;

  stage(0, 0);
  wait_async0();
  __syncthreads();

  for (int kb = 0; kb < ntiles; ++kb) {
    const int tk  = kb * 64;
    const int cur = kb & 1;
    if (kb + 1 < ntiles) stage((kb + 1) & 1, tk + 64);

    if (tk <= qRow0 + 15) {              // wave-uniform causal skip
      const unsigned short* pk = lsK + cur * 4096;
      const unsigned short* pv = lsV + cur * 4096;

      v8f s[4] = {};
#pragma unroll
      for (int ni = 0; ni < 4; ++ni)
#pragma unroll
        for (int j = 0; j < 2; ++j) {
          const unsigned short* p = pk + (ni * 16 + lane15) * 64 + j * 32 + laneHi * 8;
          s[ni] = wmma_bf16(qf[j], make_frag(p, p + 16), s[ni]);
        }

      float pr[4][8], rmax[8];
#pragma unroll
      for (int r = 0; r < 8; ++r) {
        int grow = qRow0 + r + 8 * laneHi;
        float m = NEG_INF;
#pragma unroll
        for (int ni = 0; ni < 4; ++ni) {
          int c = tk + ni * 16 + lane15;
          float v = (c <= grow) ? s[ni][r] * cs : NEG_INF;
          pr[ni][r] = v;
          m = fmaxf(m, v);
        }
        rmax[r] = m;
      }
#pragma unroll
      for (int off = 1; off < 16; off <<= 1)
#pragma unroll
        for (int r = 0; r < 8; ++r)
          rmax[r] = fmaxf(rmax[r], __shfl_xor(rmax[r], off, 32));

      float rsum[8];
#pragma unroll
      for (int r = 0; r < 8; ++r) {
        float mnew = fmaxf(mst[r], rmax[r]);
        float alpha = (mst[r] == mnew) ? 1.f : exp2f(mst[r] - mnew);
        mst[r] = mnew;
        float s4 = 0.f;
#pragma unroll
        for (int ni = 0; ni < 4; ++ni) {
          float pv = (pr[ni][r] == NEG_INF) ? 0.f : exp2f(pr[ni][r] - mnew);
          pr[ni][r] = pv;
          s4 += pv;
        }
        rsum[r] = s4;
        lst[r] *= alpha;
#pragma unroll
        for (int ni = 0; ni < 4; ++ni) o[ni][r] *= alpha;
      }
#pragma unroll
      for (int off = 1; off < 16; off <<= 1)
#pragma unroll
        for (int r = 0; r < 8; ++r)
          rsum[r] += __shfl_xor(rsum[r], off, 32);
#pragma unroll
      for (int r = 0; r < 8; ++r) lst[r] += rsum[r];

      // P: D-layout -> A-layout via per-wave LDS patch (same-wave DS in-order)
#pragma unroll
      for (int r = 0; r < 8; ++r) {
        int row = r + 8 * laneHi;
#pragma unroll
        for (int ni = 0; ni < 4; ++ni)
          pw[row * 64 + ni * 16 + lane15] = f2bf(pr[ni][r]);
      }
      v16bf pf[2];
#pragma unroll
      for (int j = 0; j < 2; ++j) {
        const unsigned short* p = pw + lane15 * 64 + j * 32 + laneHi * 8;
        pf[j] = make_frag(p, p + 16);
      }

#pragma unroll
      for (int ni = 0; ni < 4; ++ni)
#pragma unroll
        for (int j = 0; j < 2; ++j) {
          const unsigned short* p = pv + (ni * 16 + lane15) * 64 + j * 32 + laneHi * 8;
          o[ni] = wmma_bf16(pf[j], make_frag(p, p + 16), o[ni]);
        }
    }

    wait_async0();
    __syncthreads();
  }

  float inv[8];
#pragma unroll
  for (int r = 0; r < 8; ++r) inv[r] = 1.0f / lst[r];
#pragma unroll
  for (int ni = 0; ni < 4; ++ni) {
    int d = ni * 16 + lane15;
#pragma unroll
    for (int r = 0; r < 8; ++r) {
      int gq = qRow0 + r + 8 * laneHi;
      yatt[((size_t)(b * TT + gq)) * CC + h * DD + d] = f2bf(o[ni][r] * inv[r]);
    }
  }
}

// ---------------- launcher ----------------
extern "C" void kernel_launch(void* const* d_in, const int* in_sizes, int n_in,
                              void* d_out, int out_size, void* d_ws, size_t ws_size,
                              hipStream_t stream) {
  const float* x  = (const float*)d_in[0];
  const float* Wq = (const float*)d_in[1];
  const float* bq = (const float*)d_in[2];
  const float* Wk = (const float*)d_in[3];
  const float* bk = (const float*)d_in[4];
  const float* Wv = (const float*)d_in[5];
  const float* bv = (const float*)d_in[6];
  const float* Wo = (const float*)d_in[7];
  const float* bo = (const float*)d_in[8];

  char* ws = (char*)d_ws;
  const size_t SZ_X = (size_t)MM * CC * 2;
  const size_t SZ_W = (size_t)CC * CC * 2;
  unsigned short* xb   = (unsigned short*)(ws);
  unsigned short* wqb  = (unsigned short*)(ws + SZ_X);
  unsigned short* wkb  = (unsigned short*)(ws + SZ_X + SZ_W);
  unsigned short* wvb  = (unsigned short*)(ws + SZ_X + 2 * SZ_W);
  unsigned short* wob  = (unsigned short*)(ws + SZ_X + 3 * SZ_W);
  unsigned short* qb   = (unsigned short*)(ws + SZ_X + 4 * SZ_W);
  unsigned short* kb   = (unsigned short*)(ws + 2 * SZ_X + 4 * SZ_W);
  unsigned short* vtb  = (unsigned short*)(ws + 3 * SZ_X + 4 * SZ_W);
  unsigned short* yatt = (unsigned short*)(ws + 4 * SZ_X + 4 * SZ_W);

  cast_f32_bf16<<<(MM * CC) / 256, 256, 0, stream>>>(x, xb, MM * CC);
  cast_f32_bf16<<<(CC * CC) / 256, 256, 0, stream>>>(Wq, wqb, CC * CC);
  cast_f32_bf16<<<(CC * CC) / 256, 256, 0, stream>>>(Wk, wkb, CC * CC);
  cast_f32_bf16<<<(CC * CC) / 256, 256, 0, stream>>>(Wv, wvb, CC * CC);
  cast_f32_bf16<<<(CC * CC) / 256, 256, 0, stream>>>(Wo, wob, CC * CC);

  dim3 gg(CC / 128, MM / 128);  // (8, 32)
  gemm_bf16<0><<<gg, 256, 0, stream>>>(xb, wqb, bq, qb,  CC, CC, MM);  // Q
  gemm_bf16<0><<<gg, 256, 0, stream>>>(xb, wkb, bk, kb,  CC, CC, MM);  // K
  gemm_bf16<1><<<gg, 256, 0, stream>>>(xb, wvb, bv, vtb, CC, CC, MM);  // V^T

  dim3 ga(TT / 128, BB * HH);   // (16, 32)
  attn_kernel<<<ga, 256, 0, stream>>>(qb, kb, vtb, yatt);

  gemm_bf16<2><<<gg, 256, 0, stream>>>(yatt, wob, bo, d_out, CC, CC, MM);
}
